// BEVRoIHead_40853728920091
// MI455X (gfx1250) — matrix operand order repaired and checked by reading, not compiled, not run on hardware
//
#include <hip/hip_runtime.h>

typedef __attribute__((ext_vector_type(16))) _Float16 v16h;
typedef __attribute__((ext_vector_type(8)))  _Float16 v8h;
typedef __attribute__((ext_vector_type(8)))  float    v8f;

#define OUTB 3
#define SRR  2
#define CCH  256
#define HH   256
#define WWD  256
#define FDIM (OUTB*OUTB*CCH)   // 2304
#define DDIM 256

// ---------------------------------------------------------------- weight cvt
__global__ __launch_bounds__(256) void cvt_f16_kernel(
    const float* __restrict__ src, _Float16* __restrict__ dst, int n) {
  int i = blockIdx.x * blockDim.x + threadIdx.x;
  if (i < n) dst[i] = (_Float16)src[i];
}

// ---------------------------------------------------------------- fm transpose
// [B,C,H,W] -> [B,H,W,C]; 32x32 (c,x) tiles staged through LDS, coalesced
// on both the read (along x) and the write (along c).
__global__ __launch_bounds__(256) void transpose_fm_kernel(
    const float* __restrict__ fm, float* __restrict__ fmT)
{
  __shared__ float tile[32][33];
  const int xt = blockIdx.x << 5;        // x tile base
  const int ct = blockIdx.y << 5;        // c tile base
  const int by = blockIdx.z;             // b*H + y
  const int b  = by >> 8;
  const int y  = by & 255;
  const int tx = threadIdx.x & 31;
  const int ty = threadIdx.x >> 5;       // 8 rows/pass, 4 passes

  #pragma unroll
  for (int i = 0; i < 4; ++i) {
    int c = ct + ty + i * 8;
    tile[ty + i * 8][tx] =
        fm[(((size_t)b * CCH + c) * HH + y) * WWD + xt + tx];
  }
  __syncthreads();
  #pragma unroll
  for (int i = 0; i < 4; ++i) {
    int x = xt + ty + i * 8;
    fmT[(((size_t)b * HH + y) * WWD + x) * CCH + ct + tx] = tile[tx][ty + i * 8];
  }
}

// ---------------------------------------------------------------- sample coords
// computes the 36 bilinear sample descriptors for roi n into LDS
__device__ inline void compute_samples(const float* __restrict__ boxes, int n,
                                       int tid, float s_w[36][4], int s_i[36][4])
{
  if (tid < 36) {
    const float gw = (51.2f - (-51.2f)) / 256.0f;   // match reference arithmetic
    float b0  = boxes[n*7+0];
    float b1  = boxes[n*7+1];
    float bh  = boxes[n*7+4];
    float bw  = boxes[n*7+5];
    float ang = boxes[n*7+6];
    float cx = (b0 + 51.2f) / gw - 0.5f;
    float cy = (b1 + 51.2f) / gw - 0.5f;
    float rw = bw / gw;
    float rh = bh / gw;
    float th = -ang;
    float ct = cosf(th), st = sinf(th);
    float bin_h = rh * (1.0f/3.0f);
    float bin_w = rw * (1.0f/3.0f);

    int s   = tid;
    int bin = s >> 2, sub = s & 3;
    int ph  = bin / 3, pw = bin % 3;
    int sy  = sub >> 1, sx = sub & 1;
    float sgy = ((float)sy + 0.5f) * 0.5f;          // 0.25 / 0.75
    float sgx = ((float)sx + 0.5f) * 0.5f;
    float yy = -rh*0.5f + ((float)ph + sgy) * bin_h;
    float xx = -rw*0.5f + ((float)pw + sgx) * bin_w;
    float y  = yy*ct - xx*st + cy;
    float x  = yy*st + xx*ct + cx;
    bool valid = (y > -1.0f) && (y < 256.0f) && (x > -1.0f) && (x < 256.0f);
    y = fminf(fmaxf(y, 0.0f), 255.0f);
    x = fminf(fmaxf(x, 0.0f), 255.0f);
    int y0 = min((int)floorf(y), 255);
    int x0 = min((int)floorf(x), 255);
    int y1 = min(y0 + 1, 255);
    int x1 = min(x0 + 1, 255);
    float ly = y - (float)y0, lx = x - (float)x0;
    float hy = 1.0f - ly,     hx = 1.0f - lx;
    float vf = valid ? 1.0f : 0.0f;
    s_w[s][0] = hy*hx*vf; s_w[s][1] = hy*lx*vf;
    s_w[s][2] = ly*hx*vf; s_w[s][3] = ly*lx*vf;
    s_i[s][0] = y0*WWD + x0; s_i[s][1] = y0*WWD + x1;
    s_i[s][2] = y1*WWD + x0; s_i[s][3] = y1*WWD + x1;
  }
}

// ---------------------------------------------------------------- RoIAlign (NHWC)
// one block per roi; 256 threads = channels; every corner read is a fully
// coalesced 128B cacheline across the wave (channels contiguous).
__global__ __launch_bounds__(256) void roi_align_rot_nhwc_kernel(
    const float* __restrict__ fmT, const float* __restrict__ boxes,
    const int* __restrict__ bidx, _Float16* __restrict__ X, int N)
{
  __shared__ float s_w[36][4];
  __shared__ int   s_i[36][4];
  const int n   = blockIdx.x;
  const int tid = threadIdx.x;
  compute_samples(boxes, n, tid, s_w, s_i);
  __syncthreads();

  const int c = tid;
  const float* base = fmT + (size_t)bidx[n] * (HH * WWD * CCH) + c;
  _Float16* xrow = X + (size_t)n * FDIM;

  #pragma unroll
  for (int bin = 0; bin < 9; ++bin) {
    float acc = 0.0f;
    #pragma unroll
    for (int sub = 0; sub < 4; ++sub) {
      int s = bin*4 + sub;
      acc += s_w[s][0]*base[(size_t)s_i[s][0]*CCH] + s_w[s][1]*base[(size_t)s_i[s][1]*CCH]
           + s_w[s][2]*base[(size_t)s_i[s][2]*CCH] + s_w[s][3]*base[(size_t)s_i[s][3]*CCH];
    }
    xrow[bin*CCH + c] = (_Float16)(acc * 0.25f);
  }
}

// ---------------------------------------------------------------- RoIAlign (NCHW fallback)
__global__ __launch_bounds__(256) void roi_align_rot_kernel(
    const float* __restrict__ fm, const float* __restrict__ boxes,
    const int* __restrict__ bidx, _Float16* __restrict__ X, int N)
{
  __shared__ float s_w[36][4];
  __shared__ int   s_i[36][4];
  const int n   = blockIdx.x;
  const int tid = threadIdx.x;
  compute_samples(boxes, n, tid, s_w, s_i);
  __syncthreads();

  const int c = tid;
  const float* fmc = fm + ((size_t)bidx[n] * CCH + c) * (size_t)(HH * WWD);
  _Float16* xrow = X + (size_t)n * FDIM;

  #pragma unroll
  for (int bin = 0; bin < 9; ++bin) {
    float acc = 0.0f;
    #pragma unroll
    for (int sub = 0; sub < 4; ++sub) {
      int s = bin*4 + sub;
      acc += s_w[s][0]*fmc[s_i[s][0]] + s_w[s][1]*fmc[s_i[s][1]]
           + s_w[s][2]*fmc[s_i[s][2]] + s_w[s][3]*fmc[s_i[s][3]];
    }
    xrow[bin*CCH + c] = (_Float16)(acc * 0.25f);
  }
}

// ---------------------------------------------------------------- WMMA GEMM (NT)
__device__ inline v8f wmma_f16(v16h a, v16h b, v8f c) {
  return __builtin_amdgcn_wmma_f32_16x16x32_f16(
      /*neg_a=*/false, a, /*neg_b=*/false, b,
      /*c_mod=*/(short)0, c, /*reuse_a=*/false, /*reuse_b=*/false);
}

// out[m][n] = relu( sum_k A[m][k] * B[n][k] ), f16 in/out, f32 accumulate.
// Wave computes a 32x32 tile; block = 8 waves.
__global__ __launch_bounds__(256) void gemm_nt_relu_f16(
    const _Float16* __restrict__ A, const _Float16* __restrict__ B,
    _Float16* __restrict__ C, int M, int N, int K)
{
  const int lane = threadIdx.x & 31;
  const int wid  = blockIdx.x * 8 + (threadIdx.x >> 5);
  const int tilesN = N >> 5;
  const int tm = wid / tilesN;
  const int tn = wid - tm * tilesN;
  const int m0 = tm << 5, n0 = tn << 5;
  if (m0 >= M) return;                       // wave-uniform: EXEC stays all-1s

  const int llo = lane & 15;
  const int lhi = lane >> 4;

  // A frag (16x32 f16): lane llo = row; K = lhi*16 + e. Two 16B loads.
  const _Float16* a0 = A + (size_t)(m0 + llo) * K + lhi*16;
  const _Float16* a1 = a0 + (size_t)16 * K;
  // B frag (32x16 f16): lane llo = col; K = lhi*16 + e. One 32B load.
  const _Float16* b0 = B + (size_t)(n0 + llo) * K + lhi*16;
  const _Float16* b1 = b0 + (size_t)16 * K;

  v8f acc00 = {}, acc01 = {}, acc10 = {}, acc11 = {};

  for (int k = 0; k < K; k += 32) {
    v16h af0, af1;
    {
      v8h lo = *(const v8h*)(a0 + k);
      v8h hi = *(const v8h*)(a0 + k + 8);
      #pragma unroll
      for (int e = 0; e < 8; ++e) { af0[e] = lo[e]; af0[e+8] = hi[e]; }
    }
    {
      v8h lo = *(const v8h*)(a1 + k);
      v8h hi = *(const v8h*)(a1 + k + 8);
      #pragma unroll
      for (int e = 0; e < 8; ++e) { af1[e] = lo[e]; af1[e+8] = hi[e]; }
    }
    v16h bf0 = *(const v16h*)(b0 + k);
    v16h bf1 = *(const v16h*)(b1 + k);

    __builtin_prefetch(a0 + k + 256, 0, 1);  // global_prefetch ahead in K
    __builtin_prefetch(b0 + k + 256, 0, 1);

    acc00 = wmma_f16(af0, bf0, acc00);
    acc01 = wmma_f16(af0, bf1, acc01);
    acc10 = wmma_f16(af1, bf0, acc10);
    acc11 = wmma_f16(af1, bf1, acc11);
  }

  // D layout: VGPR r -> row = r + lhi*8 ; lane llo -> col
  const int col0 = n0 + llo;
  const int row0 = m0 + lhi*8;
  #pragma unroll
  for (int r = 0; r < 8; ++r) {
    C[(size_t)(row0      + r) * N + col0     ] = (_Float16)fmaxf(acc00[r], 0.0f);
    C[(size_t)(row0      + r) * N + col0 + 16] = (_Float16)fmaxf(acc01[r], 0.0f);
    C[(size_t)(row0 + 16 + r) * N + col0     ] = (_Float16)fmaxf(acc10[r], 0.0f);
    C[(size_t)(row0 + 16 + r) * N + col0 + 16] = (_Float16)fmaxf(acc11[r], 0.0f);
  }
}

// ---------------------------------------------------------------- head projections
__global__ __launch_bounds__(256) void heads_kernel(
    const _Float16* __restrict__ HC, const _Float16* __restrict__ HI,
    const _Float16* __restrict__ HR,
    const float* __restrict__ wc3, const float* __restrict__ bc3,
    const float* __restrict__ wi3, const float* __restrict__ bi3,
    const float* __restrict__ wr3, const float* __restrict__ br3,
    float* __restrict__ out, int N)
{
  int t = blockIdx.x * blockDim.x + threadIdx.x;
  if (t >= N * 9) return;
  int n = t / 9, j = t % 9;
  const _Float16* h;
  const float* w;
  float bias;
  float* o;
  if (j == 0)      { h = HC + (size_t)n*DDIM; w = wc3;              bias = bc3[0];   o = out + n; }
  else if (j == 1) { h = HI + (size_t)n*DDIM; w = wi3;              bias = bi3[0];   o = out + N + n; }
  else             { h = HR + (size_t)n*DDIM; w = wr3 + (j-2)*DDIM; bias = br3[j-2];
                     o = out + 2*(size_t)N + (size_t)n*7 + (j-2); }
  float s = bias;
  #pragma unroll 8
  for (int k = 0; k < DDIM; ++k) s += (float)h[k] * w[k];
  *o = s;
}

// ---------------------------------------------------------------- launch
extern "C" void kernel_launch(void* const* d_in, const int* in_sizes, int n_in,
                              void* d_out, int out_size, void* d_ws, size_t ws_size,
                              hipStream_t stream) {
  const float* fm    = (const float*)d_in[0];
  const float* boxes = (const float*)d_in[1];
  const int*   bidx  = (const int*)  d_in[2];
  const float* w_sh1 = (const float*)d_in[3];
  const float* w_sh2 = (const float*)d_in[4];
  const float* w_c1  = (const float*)d_in[5];
  const float* w_c2  = (const float*)d_in[6];
  const float* w_c3  = (const float*)d_in[7];
  const float* b_c3  = (const float*)d_in[8];
  const float* w_i1  = (const float*)d_in[9];
  const float* w_i2  = (const float*)d_in[10];
  const float* w_i3  = (const float*)d_in[11];
  const float* b_i3  = (const float*)d_in[12];
  const float* w_r1  = (const float*)d_in[13];
  const float* w_r2  = (const float*)d_in[14];
  const float* w_r3  = (const float*)d_in[15];
  const float* b_r3  = (const float*)d_in[16];

  const int N = in_sizes[2];                       // 2048 rois
  const int B = in_sizes[0] / (CCH * HH * WWD);    // 4

  size_t off = 0;
  auto wsalloc = [&](size_t bytes) -> void* {
    void* p = (char*)d_ws + off;
    off += (bytes + 255) & ~(size_t)255;
    return p;
  };
  _Float16* X    = (_Float16*)wsalloc((size_t)N * FDIM * sizeof(_Float16));
  _Float16* Wsh1 = (_Float16*)wsalloc((size_t)DDIM * FDIM * sizeof(_Float16));
  _Float16* Wsh2 = (_Float16*)wsalloc((size_t)DDIM * DDIM * sizeof(_Float16));
  _Float16* Wc1  = (_Float16*)wsalloc((size_t)DDIM * DDIM * sizeof(_Float16));
  _Float16* Wc2  = (_Float16*)wsalloc((size_t)DDIM * DDIM * sizeof(_Float16));
  _Float16* Wi1  = (_Float16*)wsalloc((size_t)DDIM * DDIM * sizeof(_Float16));
  _Float16* Wi2  = (_Float16*)wsalloc((size_t)DDIM * DDIM * sizeof(_Float16));
  _Float16* Wr1  = (_Float16*)wsalloc((size_t)DDIM * DDIM * sizeof(_Float16));
  _Float16* Wr2  = (_Float16*)wsalloc((size_t)DDIM * DDIM * sizeof(_Float16));
  _Float16* G1   = (_Float16*)wsalloc((size_t)N * DDIM * sizeof(_Float16));
  _Float16* SH   = (_Float16*)wsalloc((size_t)N * DDIM * sizeof(_Float16));
  _Float16* C1b  = (_Float16*)wsalloc((size_t)N * DDIM * sizeof(_Float16));
  _Float16* C2b  = (_Float16*)wsalloc((size_t)N * DDIM * sizeof(_Float16));
  _Float16* I1b  = (_Float16*)wsalloc((size_t)N * DDIM * sizeof(_Float16));
  _Float16* I2b  = (_Float16*)wsalloc((size_t)N * DDIM * sizeof(_Float16));
  _Float16* R1b  = (_Float16*)wsalloc((size_t)N * DDIM * sizeof(_Float16));
  _Float16* R2b  = (_Float16*)wsalloc((size_t)N * DDIM * sizeof(_Float16));

  // transposed feature map (big): only if workspace permits
  const size_t fmT_bytes = (size_t)B * HH * WWD * CCH * sizeof(float);
  const bool useT = (off + fmT_bytes) <= ws_size;
  float* fmT = useT ? (float*)wsalloc(fmT_bytes) : nullptr;

  auto cvt = [&](const float* s, _Float16* d, int n) {
    cvt_f16_kernel<<<(n + 255) / 256, 256, 0, stream>>>(s, d, n);
  };
  cvt(w_sh1, Wsh1, DDIM * FDIM);
  cvt(w_sh2, Wsh2, DDIM * DDIM);
  cvt(w_c1,  Wc1,  DDIM * DDIM);
  cvt(w_c2,  Wc2,  DDIM * DDIM);
  cvt(w_i1,  Wi1,  DDIM * DDIM);
  cvt(w_i2,  Wi2,  DDIM * DDIM);
  cvt(w_r1,  Wr1,  DDIM * DDIM);
  cvt(w_r2,  Wr2,  DDIM * DDIM);

  if (useT) {
    dim3 tgrid(WWD / 32, CCH / 32, B * HH);
    transpose_fm_kernel<<<tgrid, 256, 0, stream>>>(fm, fmT);
    roi_align_rot_nhwc_kernel<<<N, 256, 0, stream>>>(fmT, boxes, bidx, X, N);
  } else {
    roi_align_rot_kernel<<<N, 256, 0, stream>>>(fm, boxes, bidx, X, N);
  }

  auto gemm = [&](const _Float16* A, const _Float16* Bw, _Float16* Co,
                  int M, int Nc, int K) {
    int waves = (M / 32) * (Nc / 32);
    gemm_nt_relu_f16<<<(waves + 7) / 8, 256, 0, stream>>>(A, Bw, Co, M, Nc, K);
  };
  gemm(X,   Wsh1, G1,  N, DDIM, FDIM);   // 2048 x 256 x 2304
  gemm(G1,  Wsh2, SH,  N, DDIM, DDIM);
  gemm(SH,  Wc1,  C1b, N, DDIM, DDIM);
  gemm(C1b, Wc2,  C2b, N, DDIM, DDIM);
  gemm(SH,  Wi1,  I1b, N, DDIM, DDIM);
  gemm(I1b, Wi2,  I2b, N, DDIM, DDIM);
  gemm(SH,  Wr1,  R1b, N, DDIM, DDIM);
  gemm(R1b, Wr2,  R2b, N, DDIM, DDIM);

  heads_kernel<<<(N * 9 + 255) / 256, 256, 0, stream>>>(
      C2b, I2b, R2b, w_c3, b_c3, w_i3, b_i3, w_r3, b_r3, (float*)d_out, N);
}
